// MultiHeadAttentionRoPE_25726854103506
// MI455X (gfx1250) — matrix operand order repaired and checked by reading, not compiled, not run on hardware
//
#include <hip/hip_runtime.h>

// MI455X / gfx1250, wave32. All matmul FLOPs on v_wmma_f32_16x16x32_f16.
// WMMA operands are contiguous 16B chunks per lane (global_load_b128) per
// the ISA 7.12.2 VGPR layouts:
//   A-operand: lane L row M=L&15, half h -> K = h + (h&8) + 8*(L>>4)
//   B-operand: lane L col N=L&15, half h -> same K as A; V^T/P^T pair uses K = h + 16*(L>>4)
//   C/D:       VGPR r, lane L -> M = r + 8*(L>>4), N = L&15
// Weights are pre-converted fp32->f16 once (they live in L2: 4x2MB), so the
// GEMM inner loops are pure b128-load + WMMA with no per-tile conversions.

typedef __attribute__((ext_vector_type(16))) _Float16 v16h;
typedef __attribute__((ext_vector_type(8)))  _Float16 h8;
typedef __attribute__((ext_vector_type(8)))  float    v8f;
typedef __attribute__((ext_vector_type(4)))  float    f4;

union V16 { v16h v; h8 h[2]; };

#define D_MODEL 1024
#define N_HEADS 16
#define D_K     64
#define BATCH   4
#define SEQ     2048
#define TOKENS  (BATCH * SEQ)

// ---------------------------------------------------------------------------
// Kernel W: one-shot weight conversion fp32 -> f16 (Wq,Wk,Wv,Wo).
// 8 elements/thread: 2x b128 loads, 4x cvt_pk, 1x b128 store.
// ---------------------------------------------------------------------------
__global__ __launch_bounds__(256)
void convert_weights(const float* __restrict__ Wq, const float* __restrict__ Wk,
                     const float* __restrict__ Wv, const float* __restrict__ Wo,
                     _Float16* __restrict__ wdst) {
  const float* src = blockIdx.y == 0 ? Wq : blockIdx.y == 1 ? Wk
                   : blockIdx.y == 2 ? Wv : Wo;
  _Float16* dst = wdst + (size_t)blockIdx.y * D_MODEL * D_MODEL;
  const size_t idx = ((size_t)blockIdx.x * 256 + threadIdx.x) * 8;
  const f4* s = (const f4*)(src + idx);
  f4 x0 = s[0], x1 = s[1];
  h8 o;
#pragma unroll
  for (int j = 0; j < 4; ++j) {
    o[j]     = (_Float16)x0[j];
    o[j + 4] = (_Float16)x1[j];
  }
  *(h8*)(dst + idx) = o;
}

// ---------------------------------------------------------------------------
// Kernel A: QKV projection (X @ W^T + b) + RoPE.  One wave = 16 tokens x 64
// dims = exactly one head (head uniform per wave); A loaded once per k-step,
// reused for 4 WMMAs -> 128 WMMAs per wave.  B tiles load f16 directly.
// q,k -> f16 [b][h][s][d];  v -> f16 transposed [b][h][d][s] (h8 stores).
// ---------------------------------------------------------------------------
__global__ __launch_bounds__(32)
void qkv_proj_rope(const float* __restrict__ Q, const float* __restrict__ K,
                   const float* __restrict__ V,
                   const _Float16* __restrict__ w16,
                   const float* __restrict__ bq, const float* __restrict__ bk,
                   const float* __restrict__ bv,
                   _Float16* __restrict__ q_ws, _Float16* __restrict__ k_ws,
                   _Float16* __restrict__ vT_ws) {
  const int lane = threadIdx.x;
  const int lo   = lane & 15;
  const int hi   = lane >> 4;
  const int m0   = blockIdx.x * 16;       // token tile
  const int head = blockIdx.y;            // 64-wide N stripe == one head
  const int proj = blockIdx.z;            // 0=q 1=k 2=v

  const float*    X    = proj == 0 ? Q  : (proj == 1 ? K  : V);
  const float*    bias = proj == 0 ? bq : (proj == 1 ? bk : bv);
  const _Float16* W    = w16 + (size_t)proj * D_MODEL * D_MODEL;

  const float* xrow = X + (size_t)(m0 + lo) * D_MODEL;

  v8f acc[4] = {};
  for (int k0 = 0; k0 < D_MODEL; k0 += 32) {
    // A: 16 halves from two contiguous 8-float chunks (4x b128 + 8x cvt_pk)
    v16h a;
    {
      const f4* c0 = (const f4*)(xrow + k0 + 8 * hi);
      const f4* c1 = (const f4*)(xrow + k0 + 16 + 8 * hi);
      f4 x0 = c0[0], x1 = c0[1], x2 = c1[0], x3 = c1[1];
#pragma unroll
      for (int j = 0; j < 4; ++j) {
        a[j]      = (_Float16)x0[j];
        a[j + 4]  = (_Float16)x1[j];
        a[j + 8]  = (_Float16)x2[j];
        a[j + 12] = (_Float16)x3[j];
      }
    }
#pragma unroll
    for (int t = 0; t < 4; ++t) {
      const int nrow = head * D_K + t * 16 + lo;   // W row == B column
      const _Float16* wp = W + (size_t)nrow * D_MODEL + k0 + 8 * hi;
      V16 bm;
      bm.h[0] = *(const h8*)(wp);
      bm.h[1] = *(const h8*)(wp + 16);
      acc[t] = __builtin_amdgcn_wmma_f32_16x16x32_f16(false, a, false, bm.v,
                                                      (short)0, acc[t], false, false);
    }
  }

  const int b     = m0 >> 11;             // token tiles never straddle batch
  const int sbase = (m0 & (SEQ - 1)) + (hi << 3);

  if (proj == 2) {
    // V: store transposed [d][s]; r is contiguous in s -> 16B h8 stores
#pragma unroll
    for (int t = 0; t < 4; ++t) {
      const int d = t * 16 + lo;
      const float bval = bias[head * D_K + d];
      h8 hv;
#pragma unroll
      for (int r = 0; r < 8; ++r) hv[r] = (_Float16)(acc[t][r] + bval);
      *(h8*)(vT_ws + ((size_t)(b * N_HEADS + head) * D_K + d) * SEQ + sbase) = hv;
    }
  } else {
    _Float16* dst = (proj == 0) ? q_ws : k_ws;
#pragma unroll
    for (int t = 0; t < 4; ++t) {
      const int d = t * 16 + lo;
      const float bval = bias[head * D_K + d];
      // inv_freq = 10000^(-(2i)/64), 2i = d&~1; ln(10000)=9.2103403719
      const float inv_freq = __expf(-(float)(d & ~1) * (9.210340371976184f / 64.0f));
#pragma unroll
      for (int r = 0; r < 8; ++r) {
        const int s = sbase + r;
        float val = acc[t][r] + bval;
        // rotate-half partner = adjacent column = adjacent lane, same VGPR
        const float partner = __shfl_xor(val, 1, 32);
        const float rot = (d & 1) ? partner : -partner;
        const float ang = (float)s * inv_freq;
        val = val * __cosf(ang) + rot * __sinf(ang);
        dst[((size_t)(b * N_HEADS + head) * SEQ + s) * D_K + d] = (_Float16)val;
      }
    }
  }
}

// ---------------------------------------------------------------------------
// Kernel B: flash attention per (b,h), one wave per 16-query tile.
// S^T = K·Q^T (lane = query column -> per-lane online softmax, one
// shfl_xor(16) merge); out^T += V^T·P^T with the P^T B-operand assembled in
// registers via shfl_xor(16).  All operands: global_load_b128.
// ---------------------------------------------------------------------------
__global__ __launch_bounds__(32)
void flash_attention(const _Float16* __restrict__ q_ws,
                     const _Float16* __restrict__ k_ws,
                     const _Float16* __restrict__ vT_ws,
                     const int* __restrict__ mask,
                     _Float16* __restrict__ attn_ws) {
  const int lane = threadIdx.x;
  const int lo   = lane & 15;
  const int hi   = lane >> 4;
  const int q0   = blockIdx.x * 16;
  const int bh   = blockIdx.y;
  const int b    = bh >> 4;
  const int head = bh & 15;

  const _Float16* qbase = q_ws  + (size_t)(bh * SEQ + q0) * D_K;
  const _Float16* kbase = k_ws  + (size_t)bh * SEQ * D_K;
  const _Float16* vbase = vT_ws + (size_t)bh * D_K * SEQ;
  const int*      mk    = mask + b * SEQ;

  // Q^T B-operand (reused all chunks): 8-half chunks at k=8*hi, 16+8*hi -> 2x b128 each
  V16 bq0, bq1;
  bq0.h[0] = *(const h8*)(qbase + lo * D_K + (hi << 3));
  bq0.h[1] = *(const h8*)(qbase + lo * D_K + (hi << 3) + 16);
  bq1.h[0] = *(const h8*)(qbase + lo * D_K + 32 + (hi << 3));
  bq1.h[1] = *(const h8*)(qbase + lo * D_K + 32 + (hi << 3) + 16);

  float mrun = -3.0e38f;
  float lrun = 0.0f;
  v8f acc[4] = {};

  for (int kc = 0; kc < SEQ; kc += 32) {
    if (kc + 32 < SEQ) {                  // uniform speculative prefetch
      __builtin_prefetch(kbase + (size_t)(kc + 32) * D_K, 0, 0);
      __builtin_prefetch(vbase + kc + 32, 0, 0);
    }
    float p0[8], p1[8];
    float cmax = -3.0e38f;
#pragma unroll
    for (int t = 0; t < 2; ++t) {
      const int kt = kc + t * 16;
      const _Float16* kr = kbase + (size_t)(kt + lo) * D_K;
      V16 a0, a1;                         // K rows as A-operand (d split 2x32)
      a0.h[0] = *(const h8*)(kr + 8 * hi);
      a0.h[1] = *(const h8*)(kr + 16 + 8 * hi);
      a1.h[0] = *(const h8*)(kr + 32 + 8 * hi);
      a1.h[1] = *(const h8*)(kr + 48 + 8 * hi);
      v8f st = {};
      st = __builtin_amdgcn_wmma_f32_16x16x32_f16(false, a0.v, false, bq0.v,
                                                  (short)0, st, false, false);
      st = __builtin_amdgcn_wmma_f32_16x16x32_f16(false, a1.v, false, bq1.v,
                                                  (short)0, st, false, false);
      float* p = t ? p1 : p0;
#pragma unroll
      for (int r = 0; r < 8; ++r) {
        const int key = kt + r + (hi << 3);
        float sc = st[r] * 0.125f;        // 1/sqrt(64)
        sc = mk[key] ? -1.0e9f : sc;      // mask True -> -1e9 (reference)
        p[r] = sc;
        cmax = fmaxf(cmax, sc);
      }
    }
    cmax = fmaxf(cmax, __shfl_xor(cmax, 16, 32));
    const float mnew  = fmaxf(mrun, cmax);
    const float scale = __expf(mrun - mnew);
    float lsum = 0.0f;
#pragma unroll
    for (int r = 0; r < 8; ++r) {
      p0[r] = __expf(p0[r] - mnew);
      p1[r] = __expf(p1[r] - mnew);
      lsum += p0[r] + p1[r];
    }
    lsum += __shfl_xor(lsum, 16, 32);
    lrun = lrun * scale + lsum;
    mrun = mnew;
#pragma unroll
    for (int t = 0; t < 4; ++t)
#pragma unroll
      for (int r = 0; r < 8; ++r) acc[t][r] *= scale;

    // P^T B-operand: low lanes need chunk keys 0..15, high lanes 16..31;
    // the missing 8 live in the partner half-wave -> shfl_xor(16).
    v16h bp;
#pragma unroll
    for (int r = 0; r < 8; ++r) {
      const float o0 = __shfl_xor(p0[r], 16, 32);
      const float o1 = __shfl_xor(p1[r], 16, 32);
      bp[r]     = (_Float16)(hi ? o1    : p0[r]);
      bp[r + 8] = (_Float16)(hi ? p1[r] : o0);
    }

    // out^T += V^T(16d x 32k) @ P^T(32k x 16q)
#pragma unroll
    for (int t = 0; t < 4; ++t) {
      const _Float16* vr = vbase + (size_t)(t * 16 + lo) * SEQ + kc;
      V16 av;
      av.h[0] = *(const h8*)(vr + 16 * hi);
      av.h[1] = *(const h8*)(vr + 16 * hi + 8);
      acc[t] = __builtin_amdgcn_wmma_f32_16x16x32_f16(false, av.v, false, bp,
                                                      (short)0, acc[t], false, false);
    }
  }

  const float inv_l = 1.0f / lrun;
  // out^T: lane = query token, VGPR r contiguous in d -> 16B h8 stores
  _Float16* obase = attn_ws + (size_t)(b * SEQ + q0 + lo) * D_MODEL + head * D_K;
#pragma unroll
  for (int t = 0; t < 4; ++t) {
    h8 o;
#pragma unroll
    for (int r = 0; r < 8; ++r) o[r] = (_Float16)(acc[t][r] * inv_l);
    *(h8*)(obase + t * 16 + (hi << 3)) = o;
  }
}

// ---------------------------------------------------------------------------
// Kernel C: output projection attn(f16) @ Wo^T(f16) + bo -> f32.
// One wave = 16 tokens x 64 output dims; pure b128 loads + WMMA inner loop.
// ---------------------------------------------------------------------------
__global__ __launch_bounds__(32)
void out_proj(const _Float16* __restrict__ attn_ws,
              const _Float16* __restrict__ wo16, const float* __restrict__ bo,
              float* __restrict__ out) {
  const int lane = threadIdx.x;
  const int lo   = lane & 15;
  const int hi   = lane >> 4;
  const int m0   = blockIdx.x * 16;
  const int n0   = blockIdx.y * 64;

  const _Float16* arow = attn_ws + (size_t)(m0 + lo) * D_MODEL;

  v8f acc[4] = {};
  for (int k0 = 0; k0 < D_MODEL; k0 += 32) {
    V16 a;
    a.h[0] = *(const h8*)(arow + k0 + 8 * hi);
    a.h[1] = *(const h8*)(arow + k0 + 16 + 8 * hi);
#pragma unroll
    for (int t = 0; t < 4; ++t) {
      const int nrow = n0 + t * 16 + lo;
      const _Float16* wp = wo16 + (size_t)nrow * D_MODEL + k0 + 8 * hi;
      V16 bm;
      bm.h[0] = *(const h8*)(wp);
      bm.h[1] = *(const h8*)(wp + 16);
      acc[t] = __builtin_amdgcn_wmma_f32_16x16x32_f16(false, a.v, false, bm.v,
                                                      (short)0, acc[t], false, false);
    }
  }
#pragma unroll
  for (int t = 0; t < 4; ++t) {
    const float bval = bo[n0 + t * 16 + lo];
#pragma unroll
    for (int r = 0; r < 8; ++r) {
      out[(size_t)(m0 + r + (hi << 3)) * D_MODEL + n0 + t * 16 + lo] =
          acc[t][r] + bval;
    }
  }
}

// ---------------------------------------------------------------------------
// Workspace map (72 MB total):
//   [0,16M)   q_ws  f16 [b][h][s][d]
//   [16,32M)  k_ws  f16 [b][h][s][d]
//   [32,48M)  vT_ws f16 [b][h][d][s]
//   [48,64M)  attn  f16 [token][d_model]
//   [64,72M)  w16   f16 Wq|Wk|Wv|Wo (2MB each)
// ---------------------------------------------------------------------------
extern "C" void kernel_launch(void* const* d_in, const int* in_sizes, int n_in,
                              void* d_out, int out_size, void* d_ws, size_t ws_size,
                              hipStream_t stream) {
  const float* Q    = (const float*)d_in[0];
  const float* K    = (const float*)d_in[1];
  const float* V    = (const float*)d_in[2];
  const int*   mask = (const int*)  d_in[3];
  const float* Wq   = (const float*)d_in[4];
  const float* bq   = (const float*)d_in[5];
  const float* Wk   = (const float*)d_in[6];
  const float* bk   = (const float*)d_in[7];
  const float* Wv   = (const float*)d_in[8];
  const float* bv   = (const float*)d_in[9];
  const float* Wo   = (const float*)d_in[10];
  const float* bo   = (const float*)d_in[11];

  char* ws = (char*)d_ws;
  const size_t per = (size_t)BATCH * N_HEADS * SEQ * D_K * 2; // 16 MB (f16)
  _Float16* q_ws    = (_Float16*)(ws);
  _Float16* k_ws    = (_Float16*)(ws + per);
  _Float16* vT_ws   = (_Float16*)(ws + 2 * per);
  _Float16* attn_ws = (_Float16*)(ws + 3 * per);
  _Float16* w16     = (_Float16*)(ws + 4 * per);  // Wq,Wk,Wv,Wo f16
  _Float16* wo16    = w16 + (size_t)3 * D_MODEL * D_MODEL;

  dim3 gW(D_MODEL * D_MODEL / (256 * 8), 4);
  convert_weights<<<gW, 256, 0, stream>>>(Wq, Wk, Wv, Wo, w16);

  dim3 gA(TOKENS / 16, N_HEADS, 3);
  qkv_proj_rope<<<gA, 32, 0, stream>>>(Q, K, V, w16, bq, bk, bv,
                                       q_ws, k_ws, vT_ws);

  dim3 gB(SEQ / 16, BATCH * N_HEADS);
  flash_attention<<<gB, 32, 0, stream>>>(q_ws, k_ws, vT_ws, mask, attn_ws);

  dim3 gC(TOKENS / 16, D_MODEL / 64);
  out_proj<<<gC, 32, 0, stream>>>(attn_ws, wo16, bo, (float*)d_out);
}
